// Attention_77764677862145
// MI455X (gfx1250) — matrix operand order repaired and checked by reading, not compiled, not run on hardware
//
#include <hip/hip_runtime.h>

// ---------------------------------------------------------------------------
// Fused attention for MI455X (gfx1250, wave32, WMMA bf16, async LDS DMA).
//   Phase 1: Q/K/V projection (fp32 -> bf16 in d_ws), double-buffered LDS.
//   Phase 2: flash attention, online base-2 softmax, bf16 WMMA, fp32 accum,
//            global_load_async_to_lds_b128 pipeline for K, reg-pipeline for V.
// All B-matrix LDS fetches use 2 x ds_load_b128 (16 contiguous bf16 per lane).
// ---------------------------------------------------------------------------

typedef __attribute__((ext_vector_type(16))) __bf16        v16bf;
typedef __attribute__((ext_vector_type(2)))  __bf16        v2bf;
typedef __attribute__((ext_vector_type(8)))  float         v8f;
typedef __attribute__((ext_vector_type(8)))  unsigned int  v8u32;

constexpr int BATCH = 4;
constexpr int SEQ   = 4096;
constexpr int DM    = 1024;   // d_model
constexpr int DQ    = 64;     // d_qkv

constexpr int VTS = 40;       // Vt/WT row stride (ushorts): 80B, 16B-aligned
constexpr int PSS = 36;       // Ps row stride (floats): 144B, 16B-aligned

// ---- native bf16 conversion (hardware v_cvt_pk_bf16_f32, RNE) ----
__device__ __forceinline__ unsigned int pack2bf(float lo, float hi) {
#if defined(__has_builtin) && __has_builtin(__builtin_amdgcn_cvt_pk_bf16_f32)
  v2bf p = __builtin_amdgcn_cvt_pk_bf16_f32(lo, hi);
  return __builtin_bit_cast(unsigned int, p);
#else
  v2bf p; p[0] = (__bf16)lo; p[1] = (__bf16)hi;
  return __builtin_bit_cast(unsigned int, p);
#endif
}
__device__ __forceinline__ unsigned short f2bf(float f) {
  __bf16 h = (__bf16)f;
  return __builtin_bit_cast(unsigned short, h);
}
__device__ __forceinline__ v16bf as_bf(v8u32 x) {
  return __builtin_bit_cast(v16bf, x);
}
__device__ __forceinline__ float fast_exp2(float x) {
#if defined(__has_builtin) && __has_builtin(__builtin_amdgcn_exp2f)
  return __builtin_amdgcn_exp2f(x);
#else
  return exp2f(x);
#endif
}

// Wave-relative LDS byte offset of a generic pointer to __shared__ memory.
__device__ __forceinline__ unsigned lds_off(const void* p) {
  return (unsigned)(size_t)p;
}
// Async DMA: 16 bytes global -> LDS per lane (ASYNCcnt-tracked).
__device__ __forceinline__ void async_g2lds_b128(const void* gptr, void* lptr) {
  unsigned long long ga = (unsigned long long)(size_t)gptr;
  unsigned la = lds_off(lptr);
  asm volatile("global_load_async_to_lds_b128 %0, %1, off"
               :: "v"(la), "v"(ga) : "memory");
}
__device__ __forceinline__ void wait_asynccnt0() {
  asm volatile("s_wait_asynccnt 0x0" ::: "memory");
}
__device__ __forceinline__ void wait_dscnt0() {
  asm volatile("s_wait_dscnt 0x0" ::: "memory");
}

// B-tile fetch: 16 contiguous bf16 at 16B-aligned LDS address -> 2x ds_load_b128
__device__ __forceinline__ v16bf load_btile(const unsigned short* p) {
  const uint4* q = (const uint4*)p;
  uint4 lo = q[0], hi = q[1];
  v8u32 bu;
  bu[0] = lo.x; bu[1] = lo.y; bu[2] = lo.z; bu[3] = lo.w;
  bu[4] = hi.x; bu[5] = hi.y; bu[6] = hi.z; bu[7] = hi.w;
  return as_bf(bu);
}

// ---------------------------------------------------------------------------
// Kernel 1: fused QKV projection.  grid = (16384/128, 3), block = 256 (8 waves)
// ---------------------------------------------------------------------------
__global__ void __launch_bounds__(256)
qkv_proj_kernel(const float* __restrict__ src, const float* __restrict__ tgt,
                const float* __restrict__ wq, const float* __restrict__ bq,
                const float* __restrict__ wk, const float* __restrict__ bk,
                const float* __restrict__ wv, const float* __restrict__ bv,
                unsigned short* __restrict__ Qo,
                unsigned short* __restrict__ Ko,
                unsigned short* __restrict__ Vo)
{
  const int tid  = threadIdx.x;
  const int wave = tid >> 5;
  const int lane = tid & 31;
  const int m    = lane & 15;
  const int half = lane >> 4;

  const float* in; const float* w; const float* bias; unsigned short* out;
  if (blockIdx.y == 0)      { in = tgt; w = wq; bias = bq; out = Qo; }
  else if (blockIdx.y == 1) { in = src; w = wk; bias = bk; out = Ko; }
  else                      { in = src; w = wv; bias = bv; out = Vo; }

  __shared__ unsigned short WT[2][DQ * VTS];   // transposed bf16 weight chunk

  const int row0 = blockIdx.x * 128 + wave * 16;

  // ---- prologue: stage chunk 0 ----
  #pragma unroll
  for (int j = 0; j < 8; ++j) {
    int e  = tid + 256 * j;               // 0..2047
    int kk = e >> 6;
    int n  = e & 63;
    WT[0][n * VTS + kk] = f2bf(w[(size_t)kk * DQ + n]);
  }
  __syncthreads();

  v8f acc[4] = {};   // 16 x 64 fp32 tile

  constexpr int NK = DM / 32;
  for (int kc = 0; kc < NK; ++kc) {
    const int k0  = kc * 32;
    const int cur = kc & 1;
    const int nxt = cur ^ 1;
    const bool haveNext = (kc + 1 < NK);

    // -- issue next weight chunk global reads (coalesced) into registers --
    float wreg[8];
    if (haveNext) {
      #pragma unroll
      for (int j = 0; j < 8; ++j) {
        int e  = tid + 256 * j;
        int kk = e >> 6;
        int n  = e & 63;
        wreg[j] = w[(size_t)(k0 + 32 + kk) * DQ + n];
      }
    }

    // -- A tile (16 x 32): 4 x b128 loads, fp32 -> bf16, contiguous k = 16*half + s order --
    const float* arow = in + (size_t)(row0 + m) * DM + k0 + 16 * half;
    float4 f0 = *(const float4*)(arow);
    float4 f1 = *(const float4*)(arow + 4);
    float4 f2 = *(const float4*)(arow + 8);
    float4 f3 = *(const float4*)(arow + 12);
    v8u32 au;
    au[0] = pack2bf(f0.x, f0.y); au[1] = pack2bf(f0.z, f0.w);
    au[2] = pack2bf(f1.x, f1.y); au[3] = pack2bf(f1.z, f1.w);
    au[4] = pack2bf(f2.x, f2.y); au[5] = pack2bf(f2.z, f2.w);
    au[6] = pack2bf(f3.x, f3.y); au[7] = pack2bf(f3.z, f3.w);
    v16bf a = as_bf(au);

    // -- preload all 4 B tiles (2 x b128 each), then back-to-back WMMAs --
    v16bf bt[4];
    #pragma unroll
    for (int nt = 0; nt < 4; ++nt)
      bt[nt] = load_btile(&WT[cur][(nt * 16 + m) * VTS + 16 * half]);
    #pragma unroll
    for (int nt = 0; nt < 4; ++nt)
      acc[nt] = __builtin_amdgcn_wmma_f32_16x16x32_bf16(
          false, a, false, bt[nt], (short)0, acc[nt], false, false);
    __builtin_amdgcn_sched_group_barrier(0x100, 8, 0);   // 8 ds reads first
    __builtin_amdgcn_sched_group_barrier(0x008, 4, 0);   // then 4 WMMAs

    // -- store next weight chunk transposed (bf16) --
    if (haveNext) {
      #pragma unroll
      for (int j = 0; j < 8; ++j) {
        int e  = tid + 256 * j;
        int kk = e >> 6;
        int n  = e & 63;
        WT[nxt][n * VTS + kk] = f2bf(wreg[j]);
      }
    }
    __syncthreads();
  }

  // bias add + bf16 store
  #pragma unroll
  for (int nt = 0; nt < 4; ++nt) {
    float bsc = bias[nt * 16 + m];
    #pragma unroll
    for (int r = 0; r < 8; ++r) {
      int t = row0 + r + 8 * half;                // C-layout: M = r + 8*half
      out[(size_t)t * DQ + nt * 16 + m] = f2bf(acc[nt][r] + bsc);
    }
  }
}

// ---------------------------------------------------------------------------
// Kernel 2: flash attention.  grid = (4096/128, 4), block = 256 (8 waves).
// ---------------------------------------------------------------------------
__global__ void __launch_bounds__(256)
flash_attn_kernel(const unsigned short* __restrict__ Q,
                  const unsigned short* __restrict__ Kq,
                  const unsigned short* __restrict__ Vq,
                  const float* __restrict__ mask,
                  float* __restrict__ out)
{
  const int tid  = threadIdx.x;
  const int wave = tid >> 5;
  const int lane = tid & 31;
  const int m    = lane & 15;
  const int half = lane >> 4;
  const int b    = blockIdx.y;
  const int q0   = blockIdx.x * 128;                 // query row within batch
  const int trow = q0 + wave * 16 + m;               // this lane's query row
  const size_t base = (size_t)b * SEQ * DQ;

  // base-2 softmax: (qk + mask) * (1/8) * log2(e)
  constexpr float SCALE2 = 0.125f * 1.4426950408889634f;

  __shared__ unsigned short Ks[2][32 * DQ];    // K chunk, natural [s][d]
  __shared__ unsigned short Vt[2][DQ * VTS];   // V chunk, transposed [d][s]
  __shared__ float          Ps[8][16 * PSS];   // per-wave P transpose

  const int vsl = tid >> 3;              // V staging: key-in-chunk 0..31
  const int vd0 = (tid & 7) * 8;         // V staging: d base 0,8,...,56

  // ---- load Q A-tiles once: 4 x b128 (d 0..31, 32..63), bf16 in ws ----
  v16bf qa[2];
  {
    const unsigned short* qrow = Q + base + (size_t)trow * DQ;
    #pragma unroll
    for (int dt = 0; dt < 2; ++dt) {
      const uint4* p0 = (const uint4*)(qrow + dt * 32 + 16 * half);
      const uint4* p1 = (const uint4*)(qrow + dt * 32 + 8 + 16 * half);
      uint4 lo = p0[0], hi = p1[0];
      v8u32 au;
      au[0] = lo.x; au[1] = lo.y; au[2] = lo.z; au[3] = lo.w;
      au[4] = hi.x; au[5] = hi.y; au[6] = hi.z; au[7] = hi.w;
      qa[dt] = as_bf(au);
    }
  }

  // ---- prologue: stage chunk 0 into buffer 0 ----
  async_g2lds_b128(Kq + base + (size_t)tid * 8, &Ks[0][tid * 8]);   // 16B/lane
  {
    const unsigned int* vsrc =
        (const unsigned int*)(Vq + base + (size_t)vsl * DQ + vd0);
    unsigned int vr[4];
    #pragma unroll
    for (int j = 0; j < 4; ++j) vr[j] = vsrc[j];
    #pragma unroll
    for (int j = 0; j < 4; ++j) {
      Vt[0][(vd0 + 2 * j)     * VTS + vsl] = (unsigned short)(vr[j] & 0xFFFFu);
      Vt[0][(vd0 + 2 * j + 1) * VTS + vsl] = (unsigned short)(vr[j] >> 16);
    }
  }
  wait_asynccnt0();
  __syncthreads();

  v8f o[4] = {};                 // O accumulator, 16 x 64 fp32
  float m_run[8], l_run[8];
  #pragma unroll
  for (int r = 0; r < 8; ++r) { m_run[r] = -3.0e38f; l_run[r] = 0.0f; }

  constexpr int NCH = SEQ / 32;
  for (int sc = 0; sc < NCH; ++sc) {
    const int s0  = sc * 32;
    const int cur = sc & 1;
    const int nxt = cur ^ 1;
    const bool haveNext = (sc + 1 < NCH);

    // ---- issue next chunk's loads up front (overlap with compute) ----
    unsigned int vr[4] = {0, 0, 0, 0};
    if (haveNext) {
      const size_t s0n = (size_t)(s0 + 32) * DQ;
      async_g2lds_b128(Kq + base + s0n + (size_t)tid * 8, &Ks[nxt][tid * 8]);
      const unsigned int* vsrc =
          (const unsigned int*)(Vq + base + s0n + (size_t)vsl * DQ + vd0);
      #pragma unroll
      for (int j = 0; j < 4; ++j) vr[j] = vsrc[j];
    }

    // ---- scores = Q . K^T : 8 x b128 B loads, then 4 WMMAs ----
    v16bf kb[2][2];
    #pragma unroll
    for (int st = 0; st < 2; ++st)
      #pragma unroll
      for (int dt = 0; dt < 2; ++dt)
        kb[st][dt] = load_btile(&Ks[cur][(st * 16 + m) * DQ + dt * 32 + 16 * half]);
    v8f sacc[2] = {};
    #pragma unroll
    for (int dt = 0; dt < 2; ++dt) {
      sacc[0] = __builtin_amdgcn_wmma_f32_16x16x32_bf16(
          false, qa[dt], false, kb[0][dt], (short)0, sacc[0], false, false);
      sacc[1] = __builtin_amdgcn_wmma_f32_16x16x32_bf16(
          false, qa[dt], false, kb[1][dt], (short)0, sacc[1], false, false);
    }
    __builtin_amdgcn_sched_group_barrier(0x100, 8, 0);   // 8 ds reads first
    __builtin_amdgcn_sched_group_barrier(0x008, 4, 0);   // then 4 WMMAs

    // ---- (scores + mask) * scale, online softmax (base 2) ----
    float pv[2][8], cmax[8];
    #pragma unroll
    for (int r = 0; r < 8; ++r) {
      int t = q0 + wave * 16 + r + 8 * half;
      float v0 = (sacc[0][r] + mask[(size_t)t * SEQ + (s0 + m)])      * SCALE2;
      float v1 = (sacc[1][r] + mask[(size_t)t * SEQ + (s0 + 16 + m)]) * SCALE2;
      pv[0][r] = v0; pv[1][r] = v1;
      cmax[r]  = fmaxf(v0, v1);
    }
    #pragma unroll
    for (int r = 0; r < 8; ++r) {          // row-max across the 16-lane group
      float v = cmax[r];
      #pragma unroll
      for (int sh = 1; sh < 16; sh <<= 1) v = fmaxf(v, __shfl_xor(v, sh, 32));
      cmax[r] = v;
    }
    #pragma unroll
    for (int r = 0; r < 8; ++r) {
      float mn    = fmaxf(m_run[r], cmax[r]);
      float alpha = fast_exp2(m_run[r] - mn);
      float p0 = fast_exp2(pv[0][r] - mn);
      float p1 = fast_exp2(pv[1][r] - mn);
      pv[0][r] = p0; pv[1][r] = p1;
      float cs = p0 + p1;
      #pragma unroll
      for (int sh = 1; sh < 16; sh <<= 1) cs += __shfl_xor(cs, sh, 32);
      l_run[r] = l_run[r] * alpha + cs;
      m_run[r] = mn;
      #pragma unroll
      for (int nt = 0; nt < 4; ++nt) o[nt][r] *= alpha;   // rescale O
    }

    // ---- transpose P (C-layout -> A-layout) through per-wave LDS tile ----
    float* ps = Ps[wave];
    #pragma unroll
    for (int st = 0; st < 2; ++st)
      #pragma unroll
      for (int r = 0; r < 8; ++r)
        ps[(r + 8 * half) * PSS + st * 16 + m] = pv[st][r];
    wait_dscnt0();                          // wave-local LDS RAW fence

    // P A-tile readback: 4 x b128
    const float* pr = ps + m * PSS + 16 * half;
    float4 g0 = *(const float4*)(pr);
    float4 g1 = *(const float4*)(pr + 4);
    float4 g2 = *(const float4*)(pr + 8);
    float4 g3 = *(const float4*)(pr + 12);
    v8u32 pu;
    pu[0] = pack2bf(g0.x, g0.y); pu[1] = pack2bf(g0.z, g0.w);
    pu[2] = pack2bf(g1.x, g1.y); pu[3] = pack2bf(g1.z, g1.w);
    pu[4] = pack2bf(g2.x, g2.y); pu[5] = pack2bf(g2.z, g2.w);
    pu[6] = pack2bf(g3.x, g3.y); pu[7] = pack2bf(g3.z, g3.w);
    v16bf pa = as_bf(pu);

    // ---- O += P . V : 8 x b128 B loads, then 4 WMMAs ----
    v16bf vb[4];
    #pragma unroll
    for (int nt = 0; nt < 4; ++nt)
      vb[nt] = load_btile(&Vt[cur][(nt * 16 + m) * VTS + 16 * half]);
    #pragma unroll
    for (int nt = 0; nt < 4; ++nt)
      o[nt] = __builtin_amdgcn_wmma_f32_16x16x32_bf16(
          false, pa, false, vb[nt], (short)0, o[nt], false, false);
    __builtin_amdgcn_sched_group_barrier(0x100, 12, 0);  // P readback + vb
    __builtin_amdgcn_sched_group_barrier(0x008, 4, 0);   // then 4 WMMAs

    // ---- finish staging next chunk, single barrier ----
    if (haveNext) {
      #pragma unroll
      for (int j = 0; j < 4; ++j) {
        Vt[nxt][(vd0 + 2 * j)     * VTS + vsl] = (unsigned short)(vr[j] & 0xFFFFu);
        Vt[nxt][(vd0 + 2 * j + 1) * VTS + vsl] = (unsigned short)(vr[j] >> 16);
      }
    }
    wait_asynccnt0();
    __syncthreads();
  }

  // ---- epilogue: O * (1/l), fp32 store ----
  float inv[8];
  #pragma unroll
  for (int r = 0; r < 8; ++r) inv[r] = 1.0f / l_run[r];
  #pragma unroll
  for (int nt = 0; nt < 4; ++nt) {
    #pragma unroll
    for (int r = 0; r < 8; ++r) {
      int t = q0 + wave * 16 + r + 8 * half;
      out[((size_t)b * SEQ + t) * DQ + nt * 16 + m] = o[nt][r] * inv[r];
    }
  }
}

// ---------------------------------------------------------------------------
extern "C" void kernel_launch(void* const* d_in, const int* in_sizes, int n_in,
                              void* d_out, int out_size, void* d_ws, size_t ws_size,
                              hipStream_t stream) {
  const float* src  = (const float*)d_in[0];
  const float* tgt  = (const float*)d_in[1];
  const float* mask = (const float*)d_in[2];
  const float* wq   = (const float*)d_in[3];
  const float* bq   = (const float*)d_in[4];
  const float* wk   = (const float*)d_in[5];
  const float* bk   = (const float*)d_in[6];
  const float* wv   = (const float*)d_in[7];
  const float* bv   = (const float*)d_in[8];

  unsigned short* Q = (unsigned short*)d_ws;                 // 2 MB
  unsigned short* K = Q + (size_t)BATCH * SEQ * DQ;          // 2 MB
  unsigned short* V = K + (size_t)BATCH * SEQ * DQ;          // 2 MB

  qkv_proj_kernel<<<dim3((BATCH * SEQ) / 128, 3), 256, 0, stream>>>(
      src, tgt, wq, bq, wk, bk, wv, bv, Q, K, V);

  flash_attn_kernel<<<dim3(SEQ / 128, BATCH), 256, 0, stream>>>(
      Q, K, V, mask, (float*)d_out);
}
